// Vox_dimg_net_32246614458670
// MI455X (gfx1250) — compile-verified, matrix operand order checked by reading, not run on hardware
//
#include <hip/hip_runtime.h>

#define NPIX  3520
#define IMGH  44
#define IMGW  80
#define DBINS 50
#define FEAT  64
#define BATCH 2
#define NCAM  6
#define XD    100
#define YD    100
#define ZD    20
#define VOXN  (XD*YD*ZD)     // 200000
#define IC1   (FEAT*DBINS)   // 3200
#define OC1   256
#define IC2   256
#define OC2   64

typedef __bf16 v16bf __attribute__((ext_vector_type(16)));
typedef __bf16 v8bf  __attribute__((ext_vector_type(8)));
typedef float  v8f   __attribute__((ext_vector_type(8)));
typedef int    v4i_raw __attribute__((vector_size(16)));   // matches builtin param type

#if __has_builtin(__builtin_amdgcn_global_load_async_to_lds_b128)
#define USE_ASYNC 1
#else
#define USE_ASYNC 0
#endif

#if USE_ASYNC
// 32B per-lane fragment copy: two async b128s; INST_OFFSET applies to both
// the global and the LDS address per the async-to-LDS spec.
static __device__ __forceinline__ void async_copy_frag(const __bf16* g, v16bf* l) {
    __builtin_amdgcn_global_load_async_to_lds_b128((v4i_raw*)g, (v4i_raw*)l, 0,  0);
    __builtin_amdgcn_global_load_async_to_lds_b128((v4i_raw*)g, (v4i_raw*)l, 16, 0);
}
template <int N> static __device__ __forceinline__ void wait_async() {
#if __has_builtin(__builtin_amdgcn_s_wait_asynccnt)
    __builtin_amdgcn_s_wait_asynccnt(N);
#else
    asm volatile("s_wait_asynccnt %0" :: "i"(N) : "memory");
#endif
}
#endif

static __device__ __forceinline__ __bf16 f2bf(float f) {
    union { float f; unsigned u; } a; a.f = f;
    unsigned u = a.u;
    u += 0x7FFFu + ((u >> 16) & 1u);           // round-to-nearest-even
    union { unsigned short s; __bf16 b; } r; r.s = (unsigned short)(u >> 16);
    return r.b;
}

// ---------------------------------------------------------------------------
// 256B zero page used as the padding source for out-of-image conv taps.
// ---------------------------------------------------------------------------
__global__ void init_zpad(__bf16* z) { z[threadIdx.x] = (__bf16)0.0f; }

// ---------------------------------------------------------------------------
// Weight reorder: w [OC][IC][3][3] f32  ->  Wt [OC][9][IC] bf16
// (tap-major K: a 32-wide WMMA K-chunk stays inside one tap => contiguous
//  32B per-lane fragment loads for both A and B)
// ---------------------------------------------------------------------------
__global__ void reorder_weights(const float* __restrict__ w,
                                __bf16* __restrict__ out, int OC, int IC) {
    int t = blockIdx.x * blockDim.x + threadIdx.x;
    int total = OC * IC * 9;
    if (t >= total) return;
    int tap = t % 9;
    int c   = (t / 9) % IC;
    int m   = t / (9 * IC);
    out[((size_t)m * 9 + tap) * IC + c] = f2bf(w[t]);
}

// ---------------------------------------------------------------------------
// Projection + trilinear sampling into GEMM-ready pixel-major bf16 layout:
//   Xpf[b][pixel][f*50+d] = grid_sample(vol, project(pixel, depth_d))
// ---------------------------------------------------------------------------
__global__ void sample_trilinear(const float* __restrict__ vox,
                                 const float* __restrict__ invK,
                                 const float* __restrict__ Einv,
                                 __bf16* __restrict__ Xpf, int cam) {
    int idx = blockIdx.x * blockDim.x + threadIdx.x;
    if (idx >= BATCH * DBINS * NPIX) return;
    int p = idx % NPIX;
    int d = (idx / NPIX) % DBINS;
    int b = idx / (NPIX * DBINS);

    const float* K = invK + ((size_t)(b * NCAM + cam)) * 16;
    const float* E = Einv + ((size_t)(b * NCAM + cam)) * 16;

    float px = (float)(p % IMGW);
    float py = (float)(p / IMGW);
    float depth = 2.0f + (48.0f / 49.0f) * (float)d;   // linspace(2,50,50)

    float c0 = (K[0] * px + K[1] * py + K[2])  * depth;
    float c1 = (K[4] * px + K[5] * py + K[6])  * depth;
    float c2 = (K[8] * px + K[9] * py + K[10]) * depth;

    float ptx = E[0] * c0 + E[1] * c1 + E[2]  * c2 + E[3];
    float pty = E[4] * c0 + E[5] * c1 + E[6]  * c2 + E[7];
    float ptz = E[8] * c0 + E[9] * c1 + E[10] * c2 + E[11];

    // ((pt-str)/len*2-1 + 1)*0.5*(dim-1)  ==  (pt-str)*((dim-1)/len)
    float ix = (ptx + 50.0f);
    float iy = (pty + 50.0f);
    float iz = (ptz + 15.0f) * (19.0f / 28.5f);

    float xf = floorf(ix), yf = floorf(iy), zf = floorf(iz);
    float fx = ix - xf, fy = iy - yf, fz = iz - zf;
    int x0 = (int)xf, y0 = (int)yf, z0 = (int)zf;

    int   cidx[8];
    float cw[8];
#pragma unroll
    for (int i = 0; i < 8; ++i) {
        int dx = i & 1, dy = (i >> 1) & 1, dz = (i >> 2) & 1;
        int xi = x0 + dx, yi = y0 + dy, zi = z0 + dz;
        bool ok = (xi >= 0) & (xi < XD) & (yi >= 0) & (yi < YD) & (zi >= 0) & (zi < ZD);
        int xc = min(max(xi, 0), XD - 1);
        int yc = min(max(yi, 0), YD - 1);
        int zc = min(max(zi, 0), ZD - 1);
        cidx[i] = (zc * YD + yc) * XD + xc;
        float w = (dx ? fx : 1.0f - fx) * (dy ? fy : 1.0f - fy) * (dz ? fz : 1.0f - fz);
        cw[i] = ok ? w : 0.0f;
    }

    const float* vb = vox + (size_t)b * FEAT * VOXN;
    __bf16* xp = Xpf + ((size_t)(b * NPIX + p)) * IC1 + d;
#pragma unroll 4
    for (int f = 0; f < FEAT; ++f) {
        const float* vf = vb + (size_t)f * VOXN;
        float a = 0.0f;
#pragma unroll
        for (int i = 0; i < 8; ++i) a += cw[i] * vf[cidx[i]];
        xp[f * DBINS] = f2bf(a);
    }
}

// ---------------------------------------------------------------------------
// 3x3 conv as implicit-im2col GEMM on v_wmma_f32_16x16x32_bf16.
// Tile: M = 16*MWAVES, N = 32 (two 16-pixel sub-tiles; each wave keeps two
// accumulators sharing one A fragment -> 2 wmma per A load).
// Double-buffered LDS B staging: waves 0/1 stream the next K-chunk with
// GLOBAL_LOAD_ASYNC_TO_LDS (ASYNCcnt + s_wait_asynccnt) while all waves
// consume the current one. Out-of-image taps stream from a zero page so
// staging is branch-free and per-wave async accounting is uniform.
// ---------------------------------------------------------------------------
template <int IC, int OC, int MWAVES, bool STORE_BF16>
__global__ __launch_bounds__(MWAVES * 32)
void conv3x3_wmma(const __bf16* __restrict__ X, const __bf16* __restrict__ Wt,
                  const float* __restrict__ bias, const __bf16* __restrict__ zpad,
                  void* __restrict__ outv) {
    const int lane = threadIdx.x & 31;
    const int wave = threadIdx.x >> 5;
    const int ln   = lane & 15;   // N column (B/C/D) or M row (A)
    const int lh   = lane >> 4;   // K half-select
    const int b    = blockIdx.z;
    const int p0   = blockIdx.x << 5;                 // 32-pixel tile
    const int m0   = blockIdx.y * (MWAVES * 16) + wave * 16;

    __shared__ v16bf ldsB[2][2][32];                  // [buf][sub][slot] = 4KB

    // Staging wave w (w in {0,1}) owns sub-tile w; its lane's image column:
    const int sp   = p0 + wave * 16 + ln;             // flat pixel (valid for wave<2)
    const int spy  = sp / IMGW;
    const int spx  = sp % IMGW;

    v8f acc0, acc1;
#pragma unroll
    for (int i = 0; i < 8; ++i) { acc0[i] = 0.0f; acc1[i] = 0.0f; }

    const __bf16* Arow  = Wt + (size_t)(m0 + ln) * (9 * IC) + lh * 16;
    const size_t  xbase = (size_t)b * NPIX * IC;
    constexpr int KC    = IC / 32;

    for (int tap = 0; tap < 9; ++tap) {
        const int ty = tap / 3 - 1, tx = tap % 3 - 1;
        const __bf16* Atap = Arow + (size_t)tap * IC;

        // Branch-free staging source: real column or the zero page.
        const int  sy = spy + ty, sx = spx + tx;
        const bool valid = (sy >= 0) & (sy < IMGH) & (sx >= 0) & (sx < IMGW);
        const __bf16* Bcol = X + xbase + (size_t)(sy * IMGW + sx) * IC + lh * 16;
        const __bf16* Zcol = zpad + lh * 16;

        auto stage = [&](int kc, int buf) {
            if (wave < 2) {
                const __bf16* g = valid ? (Bcol + (size_t)kc * 32) : Zcol;
                v16bf* l = &ldsB[buf][wave][lane];
#if USE_ASYNC
                async_copy_frag(g, l);                 // 2x global_load_async_to_lds_b128
#else
                *l = *(const v16bf*)g;                 // global_load_b128 x2 + ds_store_b128 x2
#endif
            }
        };

        v16bf a_cur = *(const v16bf*)Atap;             // A chunk 0
        stage(0, 0);

        for (int kc = 0; kc < KC; ++kc) {
            const int buf = kc & 1;
            v16bf a_nxt;
            const bool more = (kc + 1) < KC;
            if (more) {
                stage(kc + 1, buf ^ 1);                // async issue, no wait
                a_nxt = *(const v16bf*)(Atap + (size_t)(kc + 1) * 32);
            }
#if USE_ASYNC
            if (wave < 2) {
                if (more) wait_async<2>();             // allow just-issued stage in flight
                else      wait_async<0>();
            }
#endif
            __syncthreads();                           // staged buf visible to all waves
            v16bf b0 = ldsB[buf][0][lane];             // ds_load_b128 x2
            v16bf b1 = ldsB[buf][1][lane];             // ds_load_b128 x2
            acc0 = __builtin_amdgcn_wmma_f32_16x16x32_bf16(
                false, a_cur, false, b0, (short)0, acc0, false, false);
            acc1 = __builtin_amdgcn_wmma_f32_16x16x32_bf16(
                false, a_cur, false, b1, (short)0, acc1, false, false);
            if (more) a_cur = a_nxt;
            __syncthreads();                           // reads done before restaging buf
        }
    }

    // C/D layout: VGPR r, lane ln/lh -> M = m0 + r + 8*lh; N = column
    const int mrow = m0 + 8 * lh;
    const int n0 = p0 + ln;        // sub-tile 0 column
    const int n1 = p0 + 16 + ln;   // sub-tile 1 column
    float bv[8];
#pragma unroll
    for (int r = 0; r < 8; ++r) bv[r] = bias[mrow + r];

    if (STORE_BF16) {
        __bf16* out = (__bf16*)outv;
        v8bf o0, o1;
#pragma unroll
        for (int r = 0; r < 8; ++r) {
            float v0 = acc0[r] + bv[r];
            float v1 = acc1[r] + bv[r];
            o0[r] = f2bf(v0 > 0.0f ? v0 : 0.0f);
            o1[r] = f2bf(v1 > 0.0f ? v1 : 0.0f);
        }
        *(v8bf*)(out + ((size_t)b * NPIX + n0) * OC + mrow) = o0;
        *(v8bf*)(out + ((size_t)b * NPIX + n1) * OC + mrow) = o1;
    } else {
        float* out = (float*)outv;
#pragma unroll
        for (int r = 0; r < 8; ++r) {
            float v0 = acc0[r] + bv[r];
            float v1 = acc1[r] + bv[r];
            size_t row = ((size_t)b * OC + mrow + r) * NPIX;
            out[row + n0] = v0 > 0.0f ? v0 : 0.0f;
            out[row + n1] = v1 > 0.0f ? v1 : 0.0f;
        }
    }
}

// ---------------------------------------------------------------------------
extern "C" void kernel_launch(void* const* d_in, const int* in_sizes, int n_in,
                              void* d_out, int out_size, void* d_ws, size_t ws_size,
                              hipStream_t stream) {
    const float* vox  = (const float*)d_in[0];
    const float* invK = (const float*)d_in[1];
    const float* Einv = (const float*)d_in[2];
    const float* w1   = (const float*)d_in[3];
    const float* b1   = (const float*)d_in[4];
    const float* w2   = (const float*)d_in[5];
    const float* b2   = (const float*)d_in[6];
    float* out = (float*)d_out;

    // Workspace carve-out (256B aligned chunks):
    //   zpad 256B | W1 bf16 [256][9][3200] | W2 bf16 [64][9][256]
    //   Xpf bf16 [B][3520][3200] (per-camera reuse) | Mid bf16 [B][3520][256]
    char* ws = (char*)d_ws;
    size_t off = 0;
    auto carve = [&](size_t bytes) {
        char* p = ws + off;
        off += (bytes + 255) & ~(size_t)255;
        return p;
    };
    __bf16* Zpad = (__bf16*)carve(256);
    __bf16* W1   = (__bf16*)carve((size_t)OC1 * 9 * IC1 * 2);
    __bf16* W2   = (__bf16*)carve((size_t)OC2 * 9 * IC2 * 2);
    __bf16* Xpf  = (__bf16*)carve((size_t)BATCH * NPIX * IC1 * 2);
    __bf16* Mid  = (__bf16*)carve((size_t)BATCH * NPIX * IC2 * 2);

    init_zpad<<<1, 128, 0, stream>>>(Zpad);
    reorder_weights<<<(OC1 * IC1 * 9 + 255) / 256, 256, 0, stream>>>(w1, W1, OC1, IC1);
    reorder_weights<<<(OC2 * IC2 * 9 + 255) / 256, 256, 0, stream>>>(w2, W2, OC2, IC2);

    for (int cam = 0; cam < NCAM; ++cam) {
        sample_trilinear<<<(BATCH * DBINS * NPIX + 255) / 256, 256, 0, stream>>>(
            vox, invK, Einv, Xpf, cam);

        // conv1: M=256 (2 blocks of 128), N=110 tiles of 32 px, K=28800
        conv3x3_wmma<IC1, OC1, 8, true>
            <<<dim3(NPIX / 32, OC1 / (8 * 16), BATCH), 8 * 32, 0, stream>>>(
                Xpf, W1, b1, Zpad, Mid);

        // conv2: M=64, K=2304, f32 plane-major out with fused bias+ReLU
        conv3x3_wmma<IC2, OC2, 4, false>
            <<<dim3(NPIX / 32, OC2 / (4 * 16), BATCH), 4 * 32, 0, stream>>>(
                Mid, W2, b2, Zpad, out + (size_t)cam * BATCH * OC2 * NPIX);
    }

    (void)in_sizes; (void)n_in; (void)out_size; (void)ws_size;
}
